// Toulmin_RNN_61125974556949
// MI455X (gfx1250) — matrix operand order repaired and checked by reading, not compiled, not run on hardware
//
#include <hip/hip_runtime.h>
#include <hip/hip_bf16.h>

// Problem constants
#define B_  64
#define S_  32
#define W_  128
#define D_  300
#define H_  512
#define CO_ 100
#define DP  320            // D padded to multiple of 32
#define KH  832            // 320 (xi) + 512 (h) concatenated K for RNN cell GEMM

typedef __attribute__((ext_vector_type(16))) __bf16        v16bf;
typedef __attribute__((ext_vector_type(8)))  float         v8f;
typedef __attribute__((ext_vector_type(8)))  unsigned short us8;
typedef __attribute__((ext_vector_type(16))) unsigned short us16;

__device__ __forceinline__ unsigned short f2bf(float f) {
  unsigned int u = __float_as_uint(f);
  u += 0x7fffu + ((u >> 16) & 1u);           // round-to-nearest-even
  return (unsigned short)(u >> 16);
}

// A fragment: lane holds row m = lane&15; K chunks [hi*8..+7] and [16+hi*8..+7]
__device__ __forceinline__ us16 ldA(const unsigned short* p, int hi) {
  us8 x = *(const us8*)(p + hi * 8);
  us8 y = *(const us8*)(p + 16 + hi * 8);
  return __builtin_shufflevector(x, y, 0,1,2,3,4,5,6,7,8,9,10,11,12,13,14,15);
}
// B fragment: lane holds col n = lane&15; K chunks [hi*8..+7] and [16+hi*8..+7]
__device__ __forceinline__ us16 ldB(const unsigned short* p, int hi) {
  us8 x = *(const us8*)(p + hi * 8);
  us8 y = *(const us8*)(p + 16 + hi * 8);
  return __builtin_shufflevector(x, y, 0,1,2,3,4,5,6,7,8,9,10,11,12,13,14,15);
}
__device__ __forceinline__ v8f wmma_bf(us16 a, us16 b, v8f c) {
  return __builtin_amdgcn_wmma_f32_16x16x32_bf16(
      false, __builtin_bit_cast(v16bf, a),
      false, __builtin_bit_cast(v16bf, b),
      (short)0, c, false, false);
}

// 2x2 output-tile GEMM over a contiguous K span, software-pipelined:
// fragments for K-tile kt+1 are issued before the 4 WMMAs of K-tile kt.
__device__ __forceinline__ void gemm_span(v8f& c00, v8f& c01, v8f& c10, v8f& c11,
                                          const unsigned short* pa0,
                                          const unsigned short* pa1,
                                          const unsigned short* pb0,
                                          const unsigned short* pb1,
                                          int nkt, int hi) {
  us16 a0 = ldA(pa0, hi), a1 = ldA(pa1, hi);
  us16 b0 = ldB(pb0, hi), b1 = ldB(pb1, hi);
  for (int kt = 0; kt < nkt; ++kt) {
    const int kn = (kt + 1 < nkt) ? (kt + 1) * 32 : kt * 32;
    us16 na0 = ldA(pa0 + kn, hi), na1 = ldA(pa1 + kn, hi);
    us16 nb0 = ldB(pb0 + kn, hi), nb1 = ldB(pb1 + kn, hi);
    c00 = wmma_bf(a0, b0, c00);
    c01 = wmma_bf(a0, b1, c01);
    c10 = wmma_bf(a1, b0, c10);
    c11 = wmma_bf(a1, b1, c11);
    a0 = na0; a1 = na1; b0 = nb0; b1 = nb1;
  }
}

// ---------------------------------------------------------------------------
// Kernel 0: pack all weights into bf16, B-fragment-friendly [n][Kpad] layouts.
// Conv weights padded to 128 output channels, proj to 320 rows (even tiles).
// ---------------------------------------------------------------------------
__global__ void pack_weights(const float* __restrict__ cw3, const float* __restrict__ cw4,
                             const float* __restrict__ cw5, const float* __restrict__ projw,
                             const float* __restrict__ wihf, const float* __restrict__ whhf,
                             const float* __restrict__ wihb, const float* __restrict__ whhb,
                             const float* __restrict__ initw, const float* __restrict__ clsw,
                             unsigned short* __restrict__ W3, unsigned short* __restrict__ W4,
                             unsigned short* __restrict__ W5, unsigned short* __restrict__ PJ,
                             unsigned short* __restrict__ WF, unsigned short* __restrict__ WB,
                             unsigned short* __restrict__ IW, unsigned short* __restrict__ CW) {
  const long long N3 = 128LL * 960, N4 = 128LL * 1280, N5 = 128LL * 1600;
  const long long NP = 320LL * 320, NF = 512LL * 832, NI = 512LL * 320, NC = 16LL * 1024;
  const long long total = N3 + N4 + N5 + NP + 2 * NF + NI + NC;
  for (long long i = blockIdx.x * (long long)blockDim.x + threadIdx.x; i < total;
       i += (long long)gridDim.x * blockDim.x) {
    long long r = i;
    if (r < N3) { int n = (int)(r / 960), k = (int)(r % 960), t = k / DP, d = k % DP;
      W3[r] = f2bf((n < CO_ && d < D_) ? cw3[(n * 3 + t) * D_ + d] : 0.f); continue; }
    r -= N3;
    if (r < N4) { int n = (int)(r / 1280), k = (int)(r % 1280), t = k / DP, d = k % DP;
      W4[r] = f2bf((n < CO_ && d < D_) ? cw4[(n * 4 + t) * D_ + d] : 0.f); continue; }
    r -= N4;
    if (r < N5) { int n = (int)(r / 1600), k = (int)(r % 1600), t = k / DP, d = k % DP;
      W5[r] = f2bf((n < CO_ && d < D_) ? cw5[(n * 5 + t) * D_ + d] : 0.f); continue; }
    r -= N5;
    if (r < NP) { int n = (int)(r / DP), d = (int)(r % DP);
      PJ[r] = f2bf((n < D_ && d < D_) ? projw[n * D_ + d] : 0.f); continue; }
    r -= NP;
    if (r < NF) { int n = (int)(r / KH), k = (int)(r % KH);
      float v = (k < DP) ? ((k < D_) ? wihf[n * D_ + k] : 0.f) : whhf[n * H_ + (k - DP)];
      WF[r] = f2bf(v); continue; }
    r -= NF;
    if (r < NF) { int n = (int)(r / KH), k = (int)(r % KH);
      float v = (k < DP) ? ((k < D_) ? wihb[n * D_ + k] : 0.f) : whhb[n * H_ + (k - DP)];
      WB[r] = f2bf(v); continue; }
    r -= NF;
    if (r < NI) { int n = (int)(r / DP), d = (int)(r % DP);
      IW[r] = f2bf((d < D_) ? initw[n * D_ + d] : 0.f); continue; }
    r -= NI;
    { int n = (int)(r / 1024), k = (int)(r % 1024);
      CW[r] = f2bf((n < 5) ? clsw[n * 1024 + k] : 0.f); }
  }
}

// ---------------------------------------------------------------------------
// Kernel 1: embedding gather + TextCNN implicit GEMM + fused bias/ReLU/maxpool
// ---------------------------------------------------------------------------
__device__ __forceinline__ void conv_job(const unsigned short* es, const unsigned short* W,
                                         const float* bias, int taps, int Pk,
                                         int mtp, int ntp, int lr, int hi, int* mxk) {
  const int RSB = taps * DP;
  const int n0 = (ntp * 2) * 16 + lr, n1 = n0 + 16;
  const unsigned short* pb0 = W + (size_t)n0 * RSB;
  const unsigned short* pb1 = W + (size_t)n1 * RSB;
  const int p0 = (mtp * 2) * 16 + lr, p1 = p0 + 16;
  v8f c00 = {}, c01 = {}, c10 = {}, c11 = {};
  for (int tap = 0; tap < taps; ++tap)
    gemm_span(c00, c01, c10, c11,
              es + (p0 + tap) * DP, es + (p1 + tap) * DP,
              pb0 + tap * DP, pb1 + tap * DP, 10, hi);
  const float bb0 = (n0 < CO_) ? bias[n0] : 0.f;
  const float bb1 = (n1 < CO_) ? bias[n1] : 0.f;
#pragma unroll
  for (int r = 0; r < 8; ++r) {
    const int m0 = (mtp * 2) * 16 + hi * 8 + r, m1 = m0 + 16;
    if (m0 < Pk && n0 < CO_) { float v = c00[r] + bb0; v = v > 0.f ? v : 0.f;
      atomicMax(mxk + n0, __float_as_int(v)); }
    if (m0 < Pk && n1 < CO_) { float v = c01[r] + bb1; v = v > 0.f ? v : 0.f;
      atomicMax(mxk + n1, __float_as_int(v)); }
    if (m1 < Pk && n0 < CO_) { float v = c10[r] + bb0; v = v > 0.f ? v : 0.f;
      atomicMax(mxk + n0, __float_as_int(v)); }
    if (m1 < Pk && n1 < CO_) { float v = c11[r] + bb1; v = v > 0.f ? v : 0.f;
      atomicMax(mxk + n1, __float_as_int(v)); }
  }
}

__global__ __launch_bounds__(256)
void cnn_kernel(const int* __restrict__ toks, const float* __restrict__ emb,
                const float* __restrict__ b3, const float* __restrict__ b4,
                const float* __restrict__ b5,
                const unsigned short* __restrict__ W3, const unsigned short* __restrict__ W4,
                const unsigned short* __restrict__ W5,
                unsigned short* __restrict__ sent_b) {
  __shared__ unsigned short es[132 * DP];   // bf16 embeddings, rows 128..131 zero
  __shared__ int mx[3 * 128];               // max-pool accumulators (float bits, >=0)
  const int tid = threadIdx.x, sid = blockIdx.x;
  for (int i = tid; i < 3 * 128; i += 256) mx[i] = 0;
  for (int i = tid; i < 132 * DP; i += 256) {
    int row = i / DP, d = i - row * DP;
    float v = 0.f;
    if (row < W_ && d < D_) {
      int tk = toks[sid * W_ + row];
      v = emb[(long long)tk * D_ + d];
    }
    es[i] = f2bf(v);
  }
  __syncthreads();

  const int lane = tid & 31, wv = tid >> 5;
  const int lr = lane & 15, hi = lane >> 4;

  // 48 macro-tiles: 3 widths x 4 M-pairs x 4 N-pairs (2x2 tiles each)
  for (int job = wv; job < 48; job += 8) {
    const int kidx = job / 16, rp = job % 16, mtp = rp >> 2, ntp = rp & 3;
    if (kidx == 0)      conv_job(es, W3, b3, 3, 126, mtp, ntp, lr, hi, mx);
    else if (kidx == 1) conv_job(es, W4, b4, 4, 125, mtp, ntp, lr, hi, mx + 128);
    else                conv_job(es, W5, b5, 5, 124, mtp, ntp, lr, hi, mx + 256);
  }
  __syncthreads();
  for (int i = tid; i < DP; i += 256) {      // write sent (bf16, K-padded)
    unsigned short v = 0;
    if (i < 3 * CO_) {
      const int kidx = i / CO_, co = i - kidx * CO_;
      v = f2bf(__int_as_float(mx[kidx * 128 + co]));
    }
    sent_b[(size_t)sid * DP + i] = v;
  }
}

// ---------------------------------------------------------------------------
// Kernel 2: bidirectional RNN scan, single block, all GEMMs via bf16 WMMA.
// ---------------------------------------------------------------------------
__device__ __forceinline__ void rnn_dir(const unsigned short* xi, const unsigned short* hbuf,
                                        const unsigned short* Wd, const float* bih,
                                        const float* bhh, unsigned short* hout,
                                        int mtp, int ntp, int lr, int hi) {
  const int n0 = ntp * 32 + lr, n1 = n0 + 16;
  const unsigned short* pb0 = Wd + (size_t)n0 * KH;
  const unsigned short* pb1 = Wd + (size_t)n1 * KH;
  const int r0 = mtp * 32 + lr, r1 = r0 + 16;
  v8f c00 = {}, c01 = {}, c10 = {}, c11 = {};
  gemm_span(c00, c01, c10, c11, xi + r0 * DP, xi + r1 * DP, pb0, pb1, 10, hi);
  gemm_span(c00, c01, c10, c11, hbuf + r0 * H_, hbuf + r1 * H_, pb0 + DP, pb1 + DP, 16, hi);
  const float bb0 = bih[n0] + bhh[n0], bb1 = bih[n1] + bhh[n1];
#pragma unroll
  for (int r = 0; r < 8; ++r) {
    const int m0 = mtp * 32 + hi * 8 + r, m1 = m0 + 16;
    hout[m0 * H_ + n0] = f2bf(tanhf(c00[r] + bb0));
    hout[m0 * H_ + n1] = f2bf(tanhf(c01[r] + bb1));
    hout[m1 * H_ + n0] = f2bf(tanhf(c10[r] + bb0));
    hout[m1 * H_ + n1] = f2bf(tanhf(c11[r] + bb1));
  }
}

__global__ __launch_bounds__(256)
void rnn_kernel(const unsigned short* __restrict__ sent_b,
                const unsigned short* __restrict__ PJ, const unsigned short* __restrict__ WF,
                const unsigned short* __restrict__ WB, const unsigned short* __restrict__ IW,
                const unsigned short* __restrict__ CW,
                const float* __restrict__ projb,
                const float* __restrict__ bihf, const float* __restrict__ bhhf,
                const float* __restrict__ bihb, const float* __restrict__ bhhb,
                const float* __restrict__ initb, const float* __restrict__ clsb,
                unsigned short* __restrict__ xi, unsigned short* __restrict__ hf,
                unsigned short* __restrict__ hb, unsigned short* __restrict__ hnf,
                unsigned short* __restrict__ hnb, float* __restrict__ out) {
  const int tid = threadIdx.x, lane = tid & 31, wv = tid >> 5;
  const int lr = lane & 15, hi = lane >> 4;

  for (int i = tid; i < B_ * DP; i += 256) xi[i] = 0;   // K-pad cols stay zero

  // h0 = sent[:,0,:] @ init_w^T + init_b   (no tanh): 2 M-pairs x 16 N-pairs
  for (int job = wv; job < 32; job += 8) {
    const int mtp = job >> 4, ntp = job & 15;
    const int n0 = ntp * 32 + lr, n1 = n0 + 16;
    const int r0 = mtp * 32 + lr, r1 = r0 + 16;
    v8f c00 = {}, c01 = {}, c10 = {}, c11 = {};
    gemm_span(c00, c01, c10, c11,
              sent_b + (size_t)(r0 * S_) * DP, sent_b + (size_t)(r1 * S_) * DP,
              IW + (size_t)n0 * DP, IW + (size_t)n1 * DP, 10, hi);
    const float bb0 = initb[n0], bb1 = initb[n1];
#pragma unroll
    for (int r = 0; r < 8; ++r) {
      const int m0 = mtp * 32 + hi * 8 + r, m1 = m0 + 16;
      unsigned short v;
      v = f2bf(c00[r] + bb0); hf[m0 * H_ + n0] = v; hb[m0 * H_ + n0] = v;
      v = f2bf(c01[r] + bb1); hf[m0 * H_ + n1] = v; hb[m0 * H_ + n1] = v;
      v = f2bf(c10[r] + bb0); hf[m1 * H_ + n0] = v; hb[m1 * H_ + n0] = v;
      v = f2bf(c11[r] + bb1); hf[m1 * H_ + n1] = v; hb[m1 * H_ + n1] = v;
    }
  }
  __threadfence(); __syncthreads();

  for (int t = 0; t < S_; ++t) {
    // Phase 1: xi = x_t @ proj_w^T + proj_b   (2 M-pairs x 10 N-pairs)
    for (int job = wv; job < 20; job += 8) {
      const int mtp = job / 10, ntp = job % 10;
      const int n0 = ntp * 32 + lr, n1 = n0 + 16;
      const int r0 = mtp * 32 + lr, r1 = r0 + 16;
      v8f c00 = {}, c01 = {}, c10 = {}, c11 = {};
      gemm_span(c00, c01, c10, c11,
                sent_b + (size_t)(r0 * S_ + t) * DP, sent_b + (size_t)(r1 * S_ + t) * DP,
                PJ + (size_t)n0 * DP, PJ + (size_t)n1 * DP, 10, hi);
      const float bb0 = (n0 < D_) ? projb[n0] : 0.f;
      const float bb1 = (n1 < D_) ? projb[n1] : 0.f;
#pragma unroll
      for (int r = 0; r < 8; ++r) {
        const int m0 = mtp * 32 + hi * 8 + r, m1 = m0 + 16;
        xi[m0 * DP + n0] = f2bf((n0 < D_) ? (c00[r] + bb0) : 0.f);
        xi[m0 * DP + n1] = f2bf((n1 < D_) ? (c01[r] + bb1) : 0.f);
        xi[m1 * DP + n0] = f2bf((n0 < D_) ? (c10[r] + bb0) : 0.f);
        xi[m1 * DP + n1] = f2bf((n1 < D_) ? (c11[r] + bb1) : 0.f);
      }
    }
    __threadfence(); __syncthreads();

    // Phase 2: h_new = tanh([xi|h] @ [w_ih;w_hh]^T + b)  (2 dirs x 2 Mp x 16 Np)
    for (int job = wv; job < 64; job += 8) {
      const int dir = job >> 5, rem = job & 31, mtp = rem >> 4, ntp = rem & 15;
      if (dir == 0) rnn_dir(xi, hf, WF, bihf, bhhf, hnf, mtp, ntp, lr, hi);
      else          rnn_dir(xi, hb, WB, bihb, bhhb, hnb, mtp, ntp, lr, hi);
    }
    __threadfence(); __syncthreads();

    // Phase 3: pred = [hf_n | hb_n] @ cls_w^T + cls_b   (K=1024, N=5 masked)
    for (int mt = wv; mt < 4; mt += 8) {
      const unsigned short* br = CW + (size_t)lr * 1024;
      const int mrow = mt * 16 + lr;
      v8f acc = {};
      for (int kt = 0; kt < 16; ++kt)
        acc = wmma_bf(ldA(hnf + mrow * H_ + kt * 32, hi), ldB(br + kt * 32, hi), acc);
      for (int kt = 0; kt < 16; ++kt)
        acc = wmma_bf(ldA(hnb + mrow * H_ + kt * 32, hi), ldB(br + 512 + kt * 32, hi), acc);
      if (lr < 5) {
        const float bb = clsb[lr];
#pragma unroll
        for (int r = 0; r < 8; ++r) {
          const int m = mt * 16 + hi * 8 + r;
          out[((size_t)m * S_ + t) * 5 + lr] = acc[r] + bb;
        }
      }
    }
    // Phase 3b: commit new hidden state
    for (int i = tid; i < B_ * H_; i += 256) { hf[i] = hnf[i]; hb[i] = hnb[i]; }
    __threadfence(); __syncthreads();
  }
}

// ---------------------------------------------------------------------------
extern "C" void kernel_launch(void* const* d_in, const int* in_sizes, int n_in,
                              void* d_out, int out_size, void* d_ws, size_t ws_size,
                              hipStream_t stream) {
  const int*   toks = (const int*)  d_in[0];
  const float* emb  = (const float*)d_in[1];
  const float* cw3  = (const float*)d_in[2];  const float* cb3 = (const float*)d_in[3];
  const float* cw4  = (const float*)d_in[4];  const float* cb4 = (const float*)d_in[5];
  const float* cw5  = (const float*)d_in[6];  const float* cb5 = (const float*)d_in[7];
  const float* pjw  = (const float*)d_in[8];  const float* pjb = (const float*)d_in[9];
  const float* iw   = (const float*)d_in[10]; const float* ib  = (const float*)d_in[11];
  const float* wihf = (const float*)d_in[12]; const float* whhf= (const float*)d_in[13];
  const float* bihf = (const float*)d_in[14]; const float* bhhf= (const float*)d_in[15];
  const float* wihb = (const float*)d_in[16]; const float* whhb= (const float*)d_in[17];
  const float* bihb = (const float*)d_in[18]; const float* bhhb= (const float*)d_in[19];
  const float* clsw = (const float*)d_in[20]; const float* clsb= (const float*)d_in[21];

  char* ws = (char*)d_ws;
  size_t off = 0;
  auto alloc = [&](size_t bytes) -> unsigned short* {
    unsigned short* p = (unsigned short*)(ws + off);
    off += (bytes + 255) & ~(size_t)255;
    return p;
  };
  unsigned short* W3  = alloc(128u * 960  * 2);
  unsigned short* W4  = alloc(128u * 1280 * 2);
  unsigned short* W5  = alloc(128u * 1600 * 2);
  unsigned short* PJ  = alloc(320u * 320  * 2);
  unsigned short* WF  = alloc(512u * 832  * 2);
  unsigned short* WB  = alloc(512u * 832  * 2);
  unsigned short* IW  = alloc(512u * 320  * 2);
  unsigned short* CW  = alloc(16u  * 1024 * 2);
  unsigned short* SB  = alloc((size_t)B_ * S_ * DP * 2);   // sent bf16
  unsigned short* XI  = alloc((size_t)B_ * DP * 2);
  unsigned short* HF  = alloc((size_t)B_ * H_ * 2);
  unsigned short* HB  = alloc((size_t)B_ * H_ * 2);
  unsigned short* HNF = alloc((size_t)B_ * H_ * 2);
  unsigned short* HNB = alloc((size_t)B_ * H_ * 2);
  (void)ws_size; (void)in_sizes; (void)n_in; (void)out_size;

  pack_weights<<<512, 256, 0, stream>>>(cw3, cw4, cw5, pjw, wihf, whhf, wihb, whhb,
                                        iw, clsw, W3, W4, W5, PJ, WF, WB, IW, CW);
  cnn_kernel<<<B_ * S_, 256, 0, stream>>>(toks, emb, cb3, cb4, cb5, W3, W4, W5, SB);
  rnn_kernel<<<1, 256, 0, stream>>>(SB, PJ, WF, WB, IW, CW, pjb,
                                    bihf, bhhf, bihb, bhhb, ib, clsb,
                                    XI, HF, HB, HNF, HNB, (float*)d_out);
}
